// MLA_20925080666179
// MI455X (gfx1250) — compile-verified, matrix operand order checked by reading, not run on hardware
//
#include <hip/hip_runtime.h>
#include <math.h>

// ---------------------------------------------------------------------------
// MLA forward for MI455X (gfx1250, wave32, WMMA).
// Shapes: B=1 S=2048 D=3072 H=24 DH=128 QP=1536 KVP=2048 NOPE=ROPE=64
// ---------------------------------------------------------------------------

typedef __attribute__((ext_vector_type(16))) __bf16 v16bf;
typedef __attribute__((ext_vector_type(8)))  __bf16 v8bf;
typedef __attribute__((ext_vector_type(8)))  float  v8f;

#define LDS_STRIDE 40   // halves per LDS tile row (32 data + 8 pad -> 80B, conflict-free b128 frag loads)

// --- optional gfx1250 async global->LDS path (guarded; falls back to ds_store)
#if defined(__gfx1250__) && __has_builtin(__builtin_amdgcn_global_load_async_to_lds_b128)
#define HAS_ASYNC_LDS 1
// Parameter types per hipcc diagnostic: pointer to GCC-style int4 vector.
typedef int v4i_g __attribute__((vector_size(16)));
typedef __attribute__((address_space(1))) v4i_g as1_v4i;   // global
typedef __attribute__((address_space(3))) v4i_g as3_v4i;   // LDS
#define ASYNC_CP16(gsrc, ldst)                                                  \
    __builtin_amdgcn_global_load_async_to_lds_b128(                             \
        (as1_v4i*)(void*)(gsrc), (as3_v4i*)(void*)(ldst), 0, 0)
#else
#define HAS_ASYNC_LDS 0
#endif

static __device__ inline void wait_async_cnt0() {
#if HAS_ASYNC_LDS
#if __has_builtin(__builtin_amdgcn_s_wait_asynccnt)
    __builtin_amdgcn_s_wait_asynccnt(0);
#else
    asm volatile("s_wait_asynccnt 0x0" ::: "memory");
#endif
#endif
}

static __device__ inline v16bf mkfrag(v8bf lo, v8bf hi) {
    v16bf f;
#pragma unroll
    for (int i = 0; i < 8; ++i) { f[i] = lo[i]; f[8 + i] = hi[i]; }
    return f;
}
static __device__ inline v8f zero8() {
    v8f z;
#pragma unroll
    for (int i = 0; i < 8; ++i) z[i] = 0.0f;
    return z;
}

// ---------------------------------------------------------------------------
// fp32 -> bf16 elementwise convert
// ---------------------------------------------------------------------------
__global__ __launch_bounds__(256) void k_convert_bf16(const float* __restrict__ in,
                                                      __bf16* __restrict__ out, int n) {
    int i = blockIdx.x * 256 + threadIdx.x;
    if (i < n) out[i] = (__bf16)in[i];
}

// ---------------------------------------------------------------------------
// fp32 (K x N) -> bf16 transposed (Npad x K), zero-fill pad rows.
// 32x32 LDS tile transpose, block (32,8).
// ---------------------------------------------------------------------------
__global__ __launch_bounds__(256) void k_transpose_bf16(const float* __restrict__ in,
                                                        __bf16* __restrict__ out,
                                                        int K, int N, int Npad) {
    __shared__ float tile[32][33];
    int bx = blockIdx.x * 32;  // n direction
    int by = blockIdx.y * 32;  // k direction
    int tx = threadIdx.x, ty = threadIdx.y;
#pragma unroll
    for (int i = ty; i < 32; i += 8) {
        int k = by + i, n = bx + tx;
        tile[i][tx] = (k < K && n < N) ? in[(size_t)k * N + n] : 0.0f;
    }
    __syncthreads();
#pragma unroll
    for (int i = ty; i < 32; i += 8) {
        int n = bx + i, k = by + tx;
        if (n < Npad && k < K) out[(size_t)n * K + k] = (__bf16)tile[tx][i];
    }
}

// ---------------------------------------------------------------------------
// bf16 WMMA GEMM: C(M x N) = A(M x K) * BT(N x K)^T, C fp32 row-major (ldc).
// Block tile 128x128, 8 waves (2 x 4), wave tile 64x32, K step 32 via LDS.
// Staging uses GLOBAL_LOAD_ASYNC_TO_LDS_B128 when available (ASYNCcnt path),
// else global_load_b128 + ds_store_b128. Next K-tile prefetched to L2.
// ---------------------------------------------------------------------------
__global__ __launch_bounds__(256) void k_gemm_bf16(const __bf16* __restrict__ A,
                                                   const __bf16* __restrict__ BT,
                                                   float* __restrict__ C,
                                                   int K, int ldc) {
    __shared__ __bf16 As[128 * LDS_STRIDE];
    __shared__ __bf16 Bs[128 * LDS_STRIDE];

    const int tid  = threadIdx.x;
    const int wave = tid >> 5;
    const int lane = tid & 31;
    const int wm = wave >> 2;       // 0..1
    const int wn = wave & 3;        // 0..3
    const int m0 = blockIdx.y * 128;
    const int n0 = blockIdx.x * 128;
    const int lm = lane & 15;       // row / col within 16
    const int hs = lane >> 4;       // half select

    v8f acc[4][2];
#pragma unroll
    for (int i = 0; i < 4; ++i)
#pragma unroll
        for (int j = 0; j < 2; ++j) acc[i][j] = zero8();

    for (int k0 = 0; k0 < K; k0 += 32) {
        // stage 128x32 bf16 tiles of A and BT (2 x 16B chunks per thread per tile)
#pragma unroll
        for (int c = tid; c < 512; c += 256) {
            int r  = c >> 2;
            int c8 = (c & 3) * 8;
            const __bf16* ga = A  + (size_t)(m0 + r) * K + k0 + c8;
            const __bf16* gb = BT + (size_t)(n0 + r) * K + k0 + c8;
#if HAS_ASYNC_LDS
            ASYNC_CP16(ga, &As[r * LDS_STRIDE + c8]);
            ASYNC_CP16(gb, &Bs[r * LDS_STRIDE + c8]);
#else
            *(v8bf*)&As[r * LDS_STRIDE + c8] = *(const v8bf*)ga;
            *(v8bf*)&Bs[r * LDS_STRIDE + c8] = *(const v8bf*)gb;
#endif
            // L2 prefetch of the next K-tile (global_prefetch_b8)
            if (k0 + 32 < K) {
                __builtin_prefetch(ga + 32, 0, 1);
                __builtin_prefetch(gb + 32, 0, 1);
            }
        }
        wait_async_cnt0();        // each wave drains its own ASYNCcnt
        __syncthreads();          // publish tile to all waves

        v16bf af[4], bfr[2];
#pragma unroll
        for (int i = 0; i < 4; ++i) {
            int r = wm * 64 + i * 16 + lm;
            v8bf lo = *(const v8bf*)&As[r * LDS_STRIDE + hs * 8];
            v8bf hi = *(const v8bf*)&As[r * LDS_STRIDE + 16 + hs * 8];
            af[i] = mkfrag(lo, hi);
        }
#pragma unroll
        for (int j = 0; j < 2; ++j) {
            int r = wn * 32 + j * 16 + lm;
            v8bf lo = *(const v8bf*)&Bs[r * LDS_STRIDE + hs * 8];
            v8bf hi = *(const v8bf*)&Bs[r * LDS_STRIDE + 16 + hs * 8];
            bfr[j] = mkfrag(lo, hi);
        }
#pragma unroll
        for (int i = 0; i < 4; ++i)
#pragma unroll
            for (int j = 0; j < 2; ++j)
                acc[i][j] = __builtin_amdgcn_wmma_f32_16x16x32_bf16(
                    false, af[i], false, bfr[j], (short)0, acc[i][j], false, false);
        __syncthreads();
    }

    // C layout: vgpr e, lane -> row = e + 8*(lane>>4), col = lane&15
#pragma unroll
    for (int i = 0; i < 4; ++i)
#pragma unroll
        for (int j = 0; j < 2; ++j)
#pragma unroll
            for (int e = 0; e < 8; ++e) {
                int r   = m0 + wm * 64 + i * 16 + e + 8 * hs;
                int col = n0 + wn * 32 + j * 16 + lm;
                C[(size_t)r * ldc + col] = acc[i][j][e];
            }
}

// ---------------------------------------------------------------------------
// block reduction helper
// ---------------------------------------------------------------------------
static __device__ inline float block_sum(float v, float* red) {
    int tid = threadIdx.x;
    red[tid] = v;
    __syncthreads();
    for (int s = 128; s > 0; s >>= 1) {
        if (tid < s) red[tid] += red[tid + s];
        __syncthreads();
    }
    float r = red[0];
    __syncthreads();
    return r;
}

// ---------------------------------------------------------------------------
// LayerNorm over `cols` fp32 -> bf16 (one block per row)
// ---------------------------------------------------------------------------
__global__ __launch_bounds__(256) void k_layernorm(const float* __restrict__ in,
                                                   const float* __restrict__ w,
                                                   const float* __restrict__ b,
                                                   __bf16* __restrict__ out, int cols) {
    __shared__ float red[256];
    int row = blockIdx.x, tid = threadIdx.x;
    const float* x = in + (size_t)row * cols;
    float s = 0.f;
    for (int i = tid; i < cols; i += 256) s += x[i];
    float mean = block_sum(s, red) / (float)cols;
    float ss = 0.f;
    for (int i = tid; i < cols; i += 256) { float d = x[i] - mean; ss += d * d; }
    float var = block_sum(ss, red) / (float)cols;
    float rs = rsqrtf(var + 1e-5f);
    for (int i = tid; i < cols; i += 256)
        out[(size_t)row * cols + i] = (__bf16)((x[i] - mean) * rs * w[i] + b[i]);
}

// ---------------------------------------------------------------------------
// compressed_kv row processing: copy fp32 row to d_out, LN first 2048 cols ->
// bf16, rope last 64 cols -> bf16 (roped K_rope shared across heads).
// ckv has padded ld = 2176.
// ---------------------------------------------------------------------------
__global__ __launch_bounds__(256) void k_kv_process(const float* __restrict__ ckv,
                                                    const float* __restrict__ w,
                                                    const float* __restrict__ b,
                                                    float* __restrict__ out_ckv,
                                                    __bf16* __restrict__ kvb,
                                                    __bf16* __restrict__ kro) {
    __shared__ float red[256];
    int row = blockIdx.x, tid = threadIdx.x;
    const float* x = ckv + (size_t)row * 2176;
    for (int i = tid; i < 2112; i += 256) out_ckv[(size_t)row * 2112 + i] = x[i];
    float s = 0.f;
    for (int i = tid; i < 2048; i += 256) s += x[i];
    float mean = block_sum(s, red) * (1.0f / 2048.0f);
    float ss = 0.f;
    for (int i = tid; i < 2048; i += 256) { float d = x[i] - mean; ss += d * d; }
    float var = block_sum(ss, red) * (1.0f / 2048.0f);
    float rs = rsqrtf(var + 1e-5f);
    for (int i = tid; i < 2048; i += 256)
        kvb[(size_t)row * 2048 + i] = (__bf16)((x[i] - mean) * rs * w[i] + b[i]);
    if (tid < 32) {
        int j = tid;
        float fr = __powf(10000.0f, -(2.0f * j) * (1.0f / 128.0f));
        float sn, cs;
        __sincosf((float)row * fr, &sn, &cs);
        float x1 = x[2048 + 2 * j], x2 = x[2048 + 2 * j + 1];
        kro[(size_t)row * 64 + j]      = (__bf16)(x1 * cs - x2 * sn);
        kro[(size_t)row * 64 + 32 + j] = (__bf16)(x1 * sn + x2 * cs);
    }
}

// ---------------------------------------------------------------------------
// Assemble q (H,S,128) bf16, k (H,S,128) bf16, vT (H,128,S) bf16.
// grid (S, H), block 128.
// ---------------------------------------------------------------------------
__global__ __launch_bounds__(128) void k_assemble(const float* __restrict__ Qm,
                                                  const float* __restrict__ KVm,
                                                  const __bf16* __restrict__ kro,
                                                  __bf16* __restrict__ qB,
                                                  __bf16* __restrict__ kB,
                                                  __bf16* __restrict__ vtB) {
    const int S = 2048;
    int s = blockIdx.x, h = blockIdx.y, d = threadIdx.x;
    size_t qo = ((size_t)h * S + s) * 128;
    // V^T: vT[h][d][s] = KV[s][h*192 + 64 + d]
    vtB[((size_t)h * 128 + d) * S + s] = (__bf16)KVm[(size_t)s * 4608 + h * 192 + 64 + d];
    if (d < 64) {
        qB[qo + d] = (__bf16)Qm[(size_t)s * 3072 + h * 128 + d];
        kB[qo + d] = (__bf16)KVm[(size_t)s * 4608 + h * 192 + d];
    } else if (d < 96) {
        int j = d - 64;
        float fr = __powf(10000.0f, -(2.0f * j) * (1.0f / 128.0f));
        float sn, cs;
        __sincosf((float)s * fr, &sn, &cs);
        float x1 = Qm[(size_t)s * 3072 + h * 128 + 64 + 2 * j];
        float x2 = Qm[(size_t)s * 3072 + h * 128 + 64 + 2 * j + 1];
        qB[qo + 64 + j] = (__bf16)(x1 * cs - x2 * sn);
        qB[qo + 96 + j] = (__bf16)(x1 * sn + x2 * cs);
        kB[qo + 64 + j] = kro[(size_t)s * 64 + j];
        kB[qo + 96 + j] = kro[(size_t)s * 64 + 32 + j];
    }
}

// ---------------------------------------------------------------------------
// Fused flash attention, causal. grid (S/128, H), block 256 (8 waves).
// Each wave owns 16 query rows; streams 64-key blocks; QK^T and PV via WMMA;
// online softmax with shfl_xor row reductions; P staged through a per-wave
// LDS slice (no cross-wave sync needed despite divergent trip counts).
// Output bf16 in (S, H*128) layout.
// ---------------------------------------------------------------------------
__global__ __launch_bounds__(256) void k_attention(const __bf16* __restrict__ qb,
                                                   const __bf16* __restrict__ kb,
                                                   const __bf16* __restrict__ vtb,
                                                   __bf16* __restrict__ ob) {
    const int S = 2048;
    const float scale = 0.08838834764831845f;  // 1/sqrt(128)
    __shared__ __bf16 plds[8][16 * 72];        // per-wave P tile, stride 72 halves

    int h    = blockIdx.y;
    int wave = threadIdx.x >> 5;
    int lane = threadIdx.x & 31;
    int qr0  = blockIdx.x * 128 + wave * 16;
    int lm   = lane & 15;
    int hs   = lane >> 4;
    __bf16* P = &plds[wave][0];

    // Q fragments (16 rows x 128 K) held in registers for the whole pass
    v16bf qf[4];
    const __bf16* qrow = qb + ((size_t)h * S + qr0 + lm) * 128;
#pragma unroll
    for (int ks = 0; ks < 4; ++ks) {
        const __bf16* p0 = qrow + ks * 32 + hs * 8;
        qf[ks] = mkfrag(*(const v8bf*)p0, *(const v8bf*)(p0 + 16));
    }

    v8f ou[8];
#pragma unroll
    for (int t = 0; t < 8; ++t) ou[t] = zero8();
    float mi[8], li[8];
#pragma unroll
    for (int j = 0; j < 8; ++j) { mi[j] = -INFINITY; li[j] = 0.0f; }

    int nkb = (qr0 + 15) / 64 + 1;
    for (int kbb = 0; kbb < nkb; ++kbb) {
        int kbase = kbb * 64;

        // L2 prefetch of next key block rows for this wave's lanes
        if (kbb + 1 < nkb) {
            const __bf16* nk = kb + ((size_t)h * S + kbase + 64 + lane) * 128;
            __builtin_prefetch(nk, 0, 1);
        }

        // --- scores: 16 x 64 tile = 4 column tiles, K = 128 in 4 steps ---
        v8f sc[4];
#pragma unroll
        for (int ct = 0; ct < 4; ++ct) {
            v8f s = zero8();
            const __bf16* krow = kb + ((size_t)h * S + kbase + ct * 16 + lm) * 128;
#pragma unroll
            for (int ks = 0; ks < 4; ++ks) {
                const __bf16* p0 = krow + ks * 32 + hs * 8;
                v16bf bfr = mkfrag(*(const v8bf*)p0, *(const v8bf*)(p0 + 16));
                s = __builtin_amdgcn_wmma_f32_16x16x32_bf16(
                    false, qf[ks], false, bfr, (short)0, s, false, false);
            }
            sc[ct] = s;
        }

        // --- scale, causal mask, row max ---
        float rmax[8];
#pragma unroll
        for (int j = 0; j < 8; ++j) rmax[j] = -INFINITY;
#pragma unroll
        for (int ct = 0; ct < 4; ++ct) {
            int col = kbase + ct * 16 + lm;
#pragma unroll
            for (int j = 0; j < 8; ++j) {
                int row = qr0 + j + 8 * hs;
                float v = (col <= row) ? sc[ct][j] * scale : -INFINITY;
                sc[ct][j] = v;
                rmax[j] = fmaxf(rmax[j], v);
            }
        }
#pragma unroll
        for (int off = 1; off < 16; off <<= 1)
#pragma unroll
            for (int j = 0; j < 8; ++j)
                rmax[j] = fmaxf(rmax[j], __shfl_xor(rmax[j], off, 32));

        // --- online softmax update ---
        float al[8], rsum[8];
#pragma unroll
        for (int j = 0; j < 8; ++j) {
            float mnew = fmaxf(mi[j], rmax[j]);
            al[j] = __expf(mi[j] - mnew);   // exp(-inf - finite) = 0 on first block
            mi[j] = mnew;
            rsum[j] = 0.0f;
        }
#pragma unroll
        for (int ct = 0; ct < 4; ++ct)
#pragma unroll
            for (int j = 0; j < 8; ++j) {
                float p = __expf(sc[ct][j] - mi[j]);
                sc[ct][j] = p;
                rsum[j] += p;
            }
#pragma unroll
        for (int off = 1; off < 16; off <<= 1)
#pragma unroll
            for (int j = 0; j < 8; ++j) rsum[j] += __shfl_xor(rsum[j], off, 32);
#pragma unroll
        for (int j = 0; j < 8; ++j) li[j] = li[j] * al[j] + rsum[j];
#pragma unroll
        for (int t = 0; t < 8; ++t)
#pragma unroll
            for (int j = 0; j < 8; ++j) ou[t][j] *= al[j];

        // --- stage P (16 x 64) to per-wave LDS as bf16 ---
#pragma unroll
        for (int ct = 0; ct < 4; ++ct)
#pragma unroll
            for (int j = 0; j < 8; ++j)
                P[(j + 8 * hs) * 72 + ct * 16 + lm] = (__bf16)sc[ct][j];

        // --- PV: out(16x128) += P(16x64) @ V(64x128), V^T streamed from global
#pragma unroll
        for (int ks2 = 0; ks2 < 2; ++ks2) {
            const __bf16* pr = P + lm * 72 + ks2 * 32 + hs * 8;
            v16bf pf = mkfrag(*(const v8bf*)pr, *(const v8bf*)(pr + 16));
#pragma unroll
            for (int ot = 0; ot < 8; ++ot) {
                const __bf16* vrow =
                    vtb + ((size_t)h * 128 + ot * 16 + lm) * S + kbase + ks2 * 32 + hs * 8;
                v16bf bfr = mkfrag(*(const v8bf*)vrow, *(const v8bf*)(vrow + 16));
                ou[ot] = __builtin_amdgcn_wmma_f32_16x16x32_bf16(
                    false, pf, false, bfr, (short)0, ou[ot], false, false);
            }
        }
    }

    // --- epilogue: divide by row sum, write bf16 (S, H*128) ---
#pragma unroll
    for (int j = 0; j < 8; ++j) li[j] = 1.0f / li[j];
#pragma unroll
    for (int ot = 0; ot < 8; ++ot)
#pragma unroll
        for (int j = 0; j < 8; ++j) {
            int row = qr0 + j + 8 * hs;
            int col = h * 128 + ot * 16 + lm;
            ob[(size_t)row * 3072 + col] = (__bf16)(ou[ot][j] * li[j]);
        }
}

// ---------------------------------------------------------------------------
// Host-side launch
// ---------------------------------------------------------------------------
extern "C" void kernel_launch(void* const* d_in, const int* in_sizes, int n_in,
                              void* d_out, int out_size, void* d_ws, size_t ws_size,
                              hipStream_t stream) {
    (void)in_sizes; (void)n_in; (void)out_size; (void)ws_size;
    const int S = 2048, D = 3072, H = 24, QP = 1536, KVP = 2048;
    const int KVW = 2112, KVWP = 2176;   // W_dkv N and padded N
    const int UKV = 4608;                // W_ukv N

    const float* x     = (const float*)d_in[0];
    const float* W_dq  = (const float*)d_in[1];
    const float* W_uq  = (const float*)d_in[2];
    const float* qlnw  = (const float*)d_in[3];
    const float* qlnb  = (const float*)d_in[4];
    const float* W_dkv = (const float*)d_in[5];
    const float* W_ukv = (const float*)d_in[6];
    const float* klnw  = (const float*)d_in[7];
    const float* klnb  = (const float*)d_in[8];
    const float* W_o   = (const float*)d_in[9];

    float* out1 = (float*)d_out;                      // (S, D) fp32
    float* out2 = out1 + (size_t)S * D;               // compressed_kv (S, 2112) fp32

    // bump allocator over workspace
    size_t off = 0;
    char* ws = (char*)d_ws;
    auto alloc = [&](size_t bytes) -> void* {
        void* p = ws + off;
        off = (off + bytes + 255) & ~(size_t)255;
        return p;
    };
    __bf16* xb     = (__bf16*)alloc((size_t)S * D * 2);
    __bf16* WdqT   = (__bf16*)alloc((size_t)QP * D * 2);
    __bf16* WuqT   = (__bf16*)alloc((size_t)D * QP * 2);
    __bf16* WdkvT  = (__bf16*)alloc((size_t)KVWP * D * 2);
    __bf16* WukvT  = (__bf16*)alloc((size_t)UKV * KVP * 2);
    __bf16* WoB    = (__bf16*)alloc((size_t)D * D * 2);
    float*  cq     = (float*) alloc((size_t)S * QP * 4);
    __bf16* cqb    = (__bf16*)alloc((size_t)S * QP * 2);
    float*  Qm     = (float*) alloc((size_t)S * D * 4);
    float*  ckv    = (float*) alloc((size_t)S * KVWP * 4);
    __bf16* kvb    = (__bf16*)alloc((size_t)S * KVP * 2);
    __bf16* kro    = (__bf16*)alloc((size_t)S * 64 * 2);
    float*  KVm    = (float*) alloc((size_t)S * UKV * 4);
    __bf16* qB     = (__bf16*)alloc((size_t)H * S * 128 * 2);
    __bf16* kB     = (__bf16*)alloc((size_t)H * S * 128 * 2);
    __bf16* vtB    = (__bf16*)alloc((size_t)H * 128 * S * 2);
    __bf16* attb   = (__bf16*)alloc((size_t)S * D * 2);

    // 1) operand conversion
    {
        int n = S * D;
        k_convert_bf16<<<(n + 255) / 256, 256, 0, stream>>>(x, xb, n);
        k_convert_bf16<<<(D * D + 255) / 256, 256, 0, stream>>>(W_o, WoB, D * D);
    }
    dim3 tb(32, 8);
    k_transpose_bf16<<<dim3(QP / 32, D / 32), tb, 0, stream>>>(W_dq, WdqT, D, QP, QP);
    k_transpose_bf16<<<dim3(D / 32, QP / 32), tb, 0, stream>>>(W_uq, WuqT, QP, D, D);
    k_transpose_bf16<<<dim3(KVWP / 32, D / 32), tb, 0, stream>>>(W_dkv, WdkvT, D, KVW, KVWP);
    k_transpose_bf16<<<dim3(UKV / 32, KVP / 32), tb, 0, stream>>>(W_ukv, WukvT, KVP, UKV, UKV);

    // 2) cq = LN(x @ W_dq)
    k_gemm_bf16<<<dim3(QP / 128, S / 128), 256, 0, stream>>>(xb, WdqT, cq, D, QP);
    k_layernorm<<<S, 256, 0, stream>>>(cq, qlnw, qlnb, cqb, QP);

    // 3) Q = cq @ W_uq
    k_gemm_bf16<<<dim3(D / 128, S / 128), 256, 0, stream>>>(cqb, WuqT, Qm, QP, D);

    // 4) compressed_kv = x @ W_dkv  (padded ldc); copy out + LN + rope K_rope
    k_gemm_bf16<<<dim3(KVWP / 128, S / 128), 256, 0, stream>>>(xb, WdkvT, ckv, D, KVWP);
    k_kv_process<<<S, 256, 0, stream>>>(ckv, klnw, klnb, out2, kvb, kro);

    // 5) KV = kv_lora @ W_ukv
    k_gemm_bf16<<<dim3(UKV / 128, S / 128), 256, 0, stream>>>(kvb, WukvT, KVm, KVP, UKV);

    // 6) assemble q/k/vT with RoPE
    k_assemble<<<dim3(S, H), 128, 0, stream>>>(Qm, KVm, kro, qB, kB, vtB);

    // 7) fused causal flash attention
    k_attention<<<dim3(S / 128, H), 256, 0, stream>>>(qB, kB, vtB, attb);

    // 8) out = attn_out @ W_o^T  -> d_out
    k_gemm_bf16<<<dim3(D / 128, S / 128), 256, 0, stream>>>(attb, WoB, out1, D, D);
}
